// Qwen3Attention_37890201485683
// MI455X (gfx1250) — compile-verified, matrix operand order checked by reading, not run on hardware
//
#include <hip/hip_runtime.h>
#include <hip/hip_bf16.h>

#define B_    2
#define S_    2048
#define HID_  1024
#define NH_   16
#define NKV_  8
#define HD_   128

typedef __attribute__((ext_vector_type(16))) __bf16    v16bf;
typedef __attribute__((ext_vector_type(8)))  float     v8f;
typedef __attribute__((ext_vector_type(4)))  unsigned  u32x4;

struct Frag32B { u32x4 a, b; };

// f32 -> bf16 (RNE) on raw bits
__device__ __forceinline__ unsigned short f2bf(float f) {
    unsigned u = __builtin_bit_cast(unsigned, f);
    unsigned r = u + 0x7FFFu + ((u >> 16) & 1u);
    return (unsigned short)(r >> 16);
}
__device__ __forceinline__ float bf2f(unsigned short h) {
    unsigned u = ((unsigned)h) << 16;
    return __builtin_bit_cast(float, u);
}

// Load one 16x32 bf16 A/B operand fragment. Source is K-contiguous rows.
// lane(0..15): row=lane, K chunks {0..7,16..23}; lane(16..31): same rows,
// K chunks {8..15,24..31}.  base = row start + k offset (elements).
__device__ __forceinline__ v16bf load_frag(const unsigned short* base, int hi) {
    Frag32B f;
    f.a = *(const u32x4*)(base + hi * 8);
    f.b = *(const u32x4*)(base + 16 + hi * 8);
    return __builtin_bit_cast(v16bf, f);
}

__device__ __forceinline__ v8f wmma_bf16(v16bf a, v16bf b, v8f c) {
    return __builtin_amdgcn_wmma_f32_16x16x32_bf16(false, a, false, b, (short)0, c, false, false);
}

// xor-lane shuffle via ds_swizzle (wave32, and_mask=0x1f)
template <int MASK>
__device__ __forceinline__ float swz(float v) {
    int i = __builtin_amdgcn_ds_swizzle(__builtin_bit_cast(int, v), (MASK << 10) | 0x1f);
    return __builtin_bit_cast(float, i);
}

// CDNA5 async global->LDS copy, 16B per lane, tracked by ASYNCcnt.
// GVS mode: lds byte addr in VDST, 32-bit per-lane offset in VADDR, SGPR64 base.
__device__ __forceinline__ void async_b128(unsigned lds, unsigned goff, unsigned long long base) {
    asm volatile("global_load_async_to_lds_b128 %0, %1, %2"
                 :: "v"(lds), "v"(goff), "s"(base) : "memory");
}
__device__ __forceinline__ void wait_async0() {
    asm volatile("s_wait_asynccnt 0" ::: "memory");
}
// Flat shared addresses carry the wave-relative LDS offset in the low 32 bits.
__device__ __forceinline__ unsigned lds_addr(const void* p) {
    return (unsigned)(uintptr_t)p;
}

// ---------------- f32 -> bf16 conversion ----------------
__global__ void cvt_kernel(const float* __restrict__ src, unsigned short* __restrict__ dst, int n) {
    int i = blockIdx.x * 256 + threadIdx.x;
    int stride = gridDim.x * 256;
    for (; i < n; i += stride) dst[i] = f2bf(src[i]);
}

// ---------------- bf16 WMMA GEMM: C[m][n] = sum_k A[m][k] * Bm[n][k] ----------------
// Block 256 thr = 8 waves (2 M x 4 N), wave tile 32x64, block tile 64x256.
// A (64x32) and B (256x32) k-step tiles staged in LDS via async copies,
// double-buffered; rows padded to 40 elems (80B) for conflict-free b128 reads.
// OUT_MODE 0: bf16 store to [B, heads, S, HD] (n = h*HD+hd, m = b*S+s)
// OUT_MODE 1: f32 store to [m][n]
template <int OUT_MODE>
__global__ __launch_bounds__(256) void gemm_bf16_kernel(
    const unsigned short* __restrict__ A,
    const unsigned short* __restrict__ Bm,
    void* __restrict__ OutP,
    int M, int N, int K, int heads)
{
    __shared__ __align__(16) unsigned short Atile[2][64 * 40];
    __shared__ __align__(16) unsigned short Btile[2][256 * 40];

    const int tid  = threadIdx.x;
    const int lane = tid & 31;
    const int wave = tid >> 5;
    const int l16  = lane & 15;
    const int hi   = lane >> 4;
    const int wm   = wave >> 2;
    const int wn   = wave & 3;
    const int mBase0 = blockIdx.x * 64;
    const int nBase0 = blockIdx.y * 256;

    // async copy assignment: thread -> (row, 16B chunk of a 64B k-row)
    const int rowT = tid >> 2;   // 0..63
    const int chk  = tid & 3;    // 0..3
    const unsigned long long aBase = (unsigned long long)(uintptr_t)A;
    const unsigned long long bBase = (unsigned long long)(uintptr_t)Bm;

    unsigned aLds[2], bLds[2];
#pragma unroll
    for (int bb = 0; bb < 2; ++bb) {
        aLds[bb] = lds_addr(&Atile[bb][rowT * 40]) + chk * 16;
        bLds[bb] = lds_addr(&Btile[bb][rowT * 40]) + chk * 16;
    }
    const unsigned aGoff0 = (unsigned)((size_t)(mBase0 + rowT) * K * 2) + chk * 16;
    unsigned bGoff0[4];
#pragma unroll
    for (int i = 0; i < 4; ++i)
        bGoff0[i] = (unsigned)((size_t)(nBase0 + rowT + 64 * i) * K * 2) + chk * 16;

    auto issue = [&](int k, int bb) {
        async_b128(aLds[bb], aGoff0 + (unsigned)k * 2, aBase);
#pragma unroll
        for (int i = 0; i < 4; ++i)
            async_b128(bLds[bb] + i * (64 * 40 * 2), bGoff0[i] + (unsigned)k * 2, bBase);
    };

    v8f acc[2][4] = {};
    issue(0, 0);
    int cur = 0;
    for (int k = 0; k < K; k += 32) {
        wait_async0();
        __syncthreads();
        if (k + 32 < K) issue(k + 32, cur ^ 1);

        const unsigned short* aT = &Atile[cur][0];
        const unsigned short* bT = &Btile[cur][0];
        v16bf af[2], bfr[4];
#pragma unroll
        for (int i = 0; i < 2; ++i)
            af[i] = load_frag(aT + (wm * 32 + 16 * i + l16) * 40, hi);
#pragma unroll
        for (int j = 0; j < 4; ++j)
            bfr[j] = load_frag(bT + (wn * 64 + 16 * j + l16) * 40, hi);
#pragma unroll
        for (int i = 0; i < 2; ++i)
#pragma unroll
            for (int j = 0; j < 4; ++j)
                acc[i][j] = wmma_bf16(af[i], bfr[j], acc[i][j]);
        cur ^= 1;
    }

    const int mBase = mBase0 + wm * 32;
    const int nBase = nBase0 + wn * 64;
    if (OUT_MODE == 0) {
        unsigned short* dst = (unsigned short*)OutP;
#pragma unroll
        for (int i = 0; i < 2; ++i)
#pragma unroll
            for (int j = 0; j < 4; ++j)
#pragma unroll
                for (int e = 0; e < 8; ++e) {
                    int row = mBase + 16 * i + e + 8 * hi;     // m = b*S + s
                    int col = nBase + 16 * j + l16;            // n = h*HD + hd
                    int bb = row >> 11, s = row & (S_ - 1);
                    int hh = col >> 7,  hd = col & (HD_ - 1);
                    dst[((size_t)(bb * heads + hh) * S_ + s) * HD_ + hd] = f2bf(acc[i][j][e]);
                }
    } else {
        float* dst = (float*)OutP;
#pragma unroll
        for (int i = 0; i < 2; ++i)
#pragma unroll
            for (int j = 0; j < 4; ++j)
#pragma unroll
                for (int e = 0; e < 8; ++e) {
                    int row = mBase + 16 * i + e + 8 * hi;
                    int col = nBase + 16 * j + l16;
                    dst[(size_t)row * N + col] = acc[i][j][e];
                }
    }
}

// ---------------- fused RMSNorm + RoPE (one 128-wide head-row per block) ----------------
__global__ __launch_bounds__(128) void norm_rope_kernel(
    const unsigned short* __restrict__ src,   // [B,H,S,HD] bf16
    unsigned short* __restrict__ dst,         // [B,H,S,HD] bf16
    const float* __restrict__ w,              // [HD]
    const float* __restrict__ cosT,           // [B,S,HD]
    const float* __restrict__ sinT,           // [B,S,HD]
    int H)
{
    __shared__ float xs[128];
    __shared__ float red[4];
    const int row = blockIdx.x;               // b*H*S + h*S + s
    const int hd  = threadIdx.x;
    const int b   = row / (H * S_);
    const int s   = row & (S_ - 1);

    float x  = bf2f(src[(size_t)row * HD_ + hd]);
    float sq = x * x;
    sq += swz<1>(sq); sq += swz<2>(sq); sq += swz<4>(sq); sq += swz<8>(sq); sq += swz<16>(sq);
    const int lane = threadIdx.x & 31, wv = threadIdx.x >> 5;
    if (lane == 0) red[wv] = sq;
    __syncthreads();
    float tot = red[0] + red[1] + red[2] + red[3];
    float inv = rsqrtf(tot * (1.0f / HD_) + 1e-6f);
    float xn  = x * inv * w[hd];
    xs[hd] = xn;
    __syncthreads();
    float rot = (hd < 64) ? -xs[hd + 64] : xs[hd - 64];
    size_t ci = ((size_t)b * S_ + s) * HD_ + hd;
    float out = xn * cosT[ci] + rot * sinT[ci];
    dst[(size_t)row * HD_ + hd] = f2bf(out);
}

// ---------------- V transpose: [B,H,S,HD] -> [B,H,HD,S] ----------------
__global__ void transpose_v_kernel(const unsigned short* __restrict__ src,
                                   unsigned short* __restrict__ dst)
{
    int idx = blockIdx.x * 256 + threadIdx.x;
    if (idx >= B_ * NKV_ * S_ * HD_) return;
    int hd = idx & (HD_ - 1);
    int t  = idx >> 7;
    int s  = t & (S_ - 1);
    int bh = t >> 11;
    dst[((size_t)bh * HD_ + hd) * S_ + s] = src[idx];
}

// ---------------- flash attention ----------------
// grid (S/64, NH, B), 128 threads = 4 waves; wave w owns q rows [q0, q0+16)
__global__ __launch_bounds__(128) void attn_kernel(
    const unsigned short* __restrict__ Q,    // [B,NH,S,HD]
    const unsigned short* __restrict__ Kb,   // [B,NKV,S,HD]
    const unsigned short* __restrict__ VT,   // [B,NKV,HD,S]
    unsigned short* __restrict__ Ctx)        // [B,S,NH*HD]
{
    __shared__ __align__(16) unsigned short pLds[4][16 * 40];  // per-wave P tile, padded
    const int lane = threadIdx.x & 31;
    const int wv   = threadIdx.x >> 5;
    const int l16  = lane & 15, hi = lane >> 4;
    const int b = blockIdx.z, h = blockIdx.y, kvh = h >> 1;    // GQA: n_rep = 2
    const int q0 = blockIdx.x * 64 + wv * 16;

    const unsigned short* qrow = Q + ((size_t)(b * NH_ + h) * S_ + q0 + l16) * HD_;
    v16bf qf[4];
#pragma unroll
    for (int d = 0; d < 4; ++d) qf[d] = load_frag(qrow + d * 32, hi);

    v8f o[8] = {};
    float mrow[8], lrow[8];
#pragma unroll
    for (int e = 0; e < 8; ++e) { mrow[e] = -3.0e38f; lrow[e] = 0.0f; }

    const unsigned short* kbase = Kb + (size_t)(b * NKV_ + kvh) * S_ * HD_;
    const unsigned short* vbase = VT + (size_t)(b * NKV_ + kvh) * HD_ * S_;
    const float scale = 0.08838834764831845f;  // HD^-0.5

    for (int kv = 0; kv <= q0 + 15; kv += 32) {
        // ---- S = Q K^T, two 16x16 n-tiles, 4 k-steps over HD ----
        v8f st[2] = {};
#pragma unroll
        for (int t = 0; t < 2; ++t) {
            const unsigned short* kr = kbase + (size_t)(kv + 16 * t + l16) * HD_;
#pragma unroll
            for (int d = 0; d < 4; ++d)
                st[t] = wmma_bf16(qf[d], load_frag(kr + d * 32, hi), st[t]);
        }
        // ---- online softmax in C-layout (row = e + 8*hi, col = 16*t + l16) ----
        float p[2][8], corr[8];
#pragma unroll
        for (int e = 0; e < 8; ++e) {
            int qg = q0 + e + 8 * hi;
            float s0 = st[0][e] * scale; if (kv +      l16 > qg) s0 = -1e30f;
            float s1 = st[1][e] * scale; if (kv + 16 + l16 > qg) s1 = -1e30f;
            float mx = fmaxf(s0, s1);
            mx = fmaxf(mx, swz<1>(mx)); mx = fmaxf(mx, swz<2>(mx));
            mx = fmaxf(mx, swz<4>(mx)); mx = fmaxf(mx, swz<8>(mx));
            float mnew = fmaxf(mrow[e], mx);
            float c = __expf(mrow[e] - mnew);
            mrow[e] = mnew; corr[e] = c;
            float p0 = __expf(s0 - mnew);
            float p1 = __expf(s1 - mnew);
            p[0][e] = p0; p[1][e] = p1;
            float rs = p0 + p1;
            rs += swz<1>(rs); rs += swz<2>(rs); rs += swz<4>(rs); rs += swz<8>(rs);
            lrow[e] = lrow[e] * c + rs;
        }
#pragma unroll
        for (int t = 0; t < 8; ++t)
#pragma unroll
            for (int e = 0; e < 8; ++e) o[t][e] *= corr[e];
        // ---- stage P through LDS to get A-fragment layout ----
#pragma unroll
        for (int t = 0; t < 2; ++t)
#pragma unroll
            for (int e = 0; e < 8; ++e)
                pLds[wv][(e + 8 * hi) * 40 + 16 * t + l16] = f2bf(p[t][e]);
        asm volatile("s_wait_dscnt 0" ::: "memory");
        v16bf pf = load_frag(&pLds[wv][l16 * 40], hi);
        // ---- O += P V  (B operand from V^T, kv-contiguous) ----
#pragma unroll
        for (int t = 0; t < 8; ++t) {
            const unsigned short* vr = vbase + (size_t)(16 * t + l16) * S_ + kv;
            o[t] = wmma_bf16(pf, load_frag(vr, hi), o[t]);
        }
    }
    // ---- epilogue: normalize and store ctx[b,s,h*HD+hd] ----
#pragma unroll
    for (int e = 0; e < 8; ++e) lrow[e] = 1.0f / lrow[e];
#pragma unroll
    for (int t = 0; t < 8; ++t)
#pragma unroll
        for (int e = 0; e < 8; ++e) {
            int s   = q0 + e + 8 * hi;
            int col = h * HD_ + 16 * t + l16;
            Ctx[((size_t)b * S_ + s) * (NH_ * HD_) + col] = f2bf(o[t][e] * lrow[e]);
        }
}

extern "C" void kernel_launch(void* const* d_in, const int* in_sizes, int n_in,
                              void* d_out, int out_size, void* d_ws, size_t ws_size,
                              hipStream_t stream)
{
    (void)in_sizes; (void)n_in; (void)out_size; (void)ws_size;
    const float* hidden = (const float*)d_in[0];
    const float* cosT   = (const float*)d_in[1];
    const float* sinT   = (const float*)d_in[2];
    // d_in[3] = attention_mask: exactly the causal -1e9 triu; applied analytically.
    const float* Wq = (const float*)d_in[4];
    const float* Wk = (const float*)d_in[5];
    const float* Wv = (const float*)d_in[6];
    const float* Wo = (const float*)d_in[7];
    const float* qw = (const float*)d_in[8];
    const float* kw = (const float*)d_in[9];

    char* base = (char*)d_ws;
    size_t off = 0;
    auto alloc = [&](size_t elems) -> unsigned short* {
        unsigned short* p = (unsigned short*)(base + off);
        off += (elems * 2 + 255) & ~(size_t)255;
        return p;
    };
    unsigned short* hB   = alloc((size_t)B_ * S_ * HID_);
    unsigned short* WqB  = alloc((size_t)NH_ * HD_ * HID_);
    unsigned short* WkB  = alloc((size_t)NKV_ * HD_ * HID_);
    unsigned short* WvB  = alloc((size_t)NKV_ * HD_ * HID_);
    unsigned short* WoB  = alloc((size_t)HID_ * NH_ * HD_);
    unsigned short* q0B  = alloc((size_t)B_ * NH_ * S_ * HD_);
    unsigned short* k0B  = alloc((size_t)B_ * NKV_ * S_ * HD_);
    unsigned short* vB   = alloc((size_t)B_ * NKV_ * S_ * HD_);
    unsigned short* qB   = alloc((size_t)B_ * NH_ * S_ * HD_);
    unsigned short* kB   = alloc((size_t)B_ * NKV_ * S_ * HD_);
    unsigned short* vtB  = alloc((size_t)B_ * NKV_ * S_ * HD_);
    unsigned short* ctxB = alloc((size_t)B_ * S_ * NH_ * HD_);

    cvt_kernel<<<4096, 256, 0, stream>>>(hidden, hB, B_ * S_ * HID_);
    cvt_kernel<<<2048, 256, 0, stream>>>(Wq, WqB, NH_ * HD_ * HID_);
    cvt_kernel<<<1024, 256, 0, stream>>>(Wk, WkB, NKV_ * HD_ * HID_);
    cvt_kernel<<<1024, 256, 0, stream>>>(Wv, WvB, NKV_ * HD_ * HID_);
    cvt_kernel<<<2048, 256, 0, stream>>>(Wo, WoB, HID_ * NH_ * HD_);

    dim3 blk(256);
    gemm_bf16_kernel<0><<<dim3(64, 8), blk, 0, stream>>>(hB, WqB, q0B, 4096, 2048, 1024, NH_);
    gemm_bf16_kernel<0><<<dim3(64, 4), blk, 0, stream>>>(hB, WkB, k0B, 4096, 1024, 1024, NKV_);
    gemm_bf16_kernel<0><<<dim3(64, 4), blk, 0, stream>>>(hB, WvB, vB,  4096, 1024, 1024, NKV_);

    norm_rope_kernel<<<B_ * NH_ * S_, 128, 0, stream>>>(q0B, qB, qw, cosT, sinT, NH_);
    norm_rope_kernel<<<B_ * NKV_ * S_, 128, 0, stream>>>(k0B, kB, kw, cosT, sinT, NKV_);
    transpose_v_kernel<<<(B_ * NKV_ * S_ * HD_) / 256, 256, 0, stream>>>(vB, vtB);

    attn_kernel<<<dim3(S_ / 64, NH_, B_), 128, 0, stream>>>(qB, kB, vtB, ctxB);

    gemm_bf16_kernel<1><<<dim3(64, 4), blk, 0, stream>>>(ctxB, WoB, d_out, 4096, 1024, 2048, 0);
}
